// MultiHeadAttention_25409026523886
// MI455X (gfx1250) — matrix operand order repaired and checked by reading, not compiled, not run on hardware
//
#include <hip/hip_runtime.h>

// MI455X / gfx1250, wave32. All matmuls via v_wmma_f32_16x16x32_f16.
// Tile staging now uses the CDNA5 async path: global_load_async_to_lds_b128
// (ASYNCcnt) instead of global_load_b128 + ds_store_b128 — no staging VGPRs,
// single s_wait_asynccnt before the workgroup barrier.
//
// Workspace layout (needs ~82 MB of d_ws):
//   hT  : groupnorm output, f16, [B][HW][C]        16 MiB
//   Qh  : [B*nh][HW][dh] f16 (scale folded in)     16 MiB
//   Kh  : [B*nh][HW][dh] f16                       16 MiB
//   Vt  : [B*nh][dh][HW] f16 (transposed for PV)   16 MiB
//   Oh  : attention out, f16, [B][HW][C]           16 MiB
//   wqh : w_qkv f16 (1536x512), wph : w_proj f16 (512x512)

typedef __attribute__((ext_vector_type(16))) _Float16 v16h;
typedef __attribute__((ext_vector_type(8)))  float    v8f;

#define WMMA(A, B, C) \
  __builtin_amdgcn_wmma_f32_16x16x32_f16(false, (A), false, (B), (short)0, (C), false, false)

// Async 16B global->LDS copy (GVS addressing: SGPR base + 32-bit VGPR offset).
// ldsaddr: LDS byte address (low 32 bits of flat pointer == addrspace(3) offset).
#define ASYNC_G2L_B128(ldsaddr, voff, sbase)                          \
  asm volatile("global_load_async_to_lds_b128 %0, %1, %2"             \
               :: "v"(ldsaddr), "v"(voff), "s"(sbase) : "memory")
#define WAIT_ASYNC() asm volatile("s_wait_asynccnt 0x0" ::: "memory")

static __device__ inline unsigned lds_off(const void* p) {
  return (unsigned)(unsigned long long)p;   // LDS aperture: low 32 bits = ds offset
}

// ---- WMMA fragment loaders (layouts per CDNA5 ISA 7.12.2) --------------------
// A (16x32 f16): lane m = lane&15, half = lane>>4; VGPR i holds
// K = (i/4)*16 + half*8 + (i%4)*2 (pair). src row-major [m][k], stride ld halves.
static __device__ inline v16h frag_a(const _Float16* src, int ld) {
  const int lane = threadIdx.x & 31;
  const int m = lane & 15, hf = lane >> 4;
  union { v16h v; unsigned u[8]; } r;
#pragma unroll
  for (int i = 0; i < 8; ++i) {
    const int k = ((i >> 2) << 4) + (hf << 3) + ((i & 3) << 1);
    r.u[i] = *(const unsigned*)(src + m * ld + k);
  }
  return r.v;
}

// B (32x16 f16): lane n = column; VGPR i holds K = (i/4)*16 + half*8 + (i%4)*2 (pair).
// src is "B-transposed" storage: row n holds that column's K-run, stride ld.
static __device__ inline v16h frag_b(const _Float16* src, int ld) {
  const int lane = threadIdx.x & 31;
  const int n = lane & 15, hf = lane >> 4;
  union { v16h v; unsigned u[8]; } r;
#pragma unroll
  for (int i = 0; i < 8; ++i) {
    const int k = ((i >> 2) << 4) + (hf << 3) + ((i & 3) << 1);
    r.u[i] = *(const unsigned*)(src + n * ld + k);
  }
  return r.v;
}

// ---- Kernel 1: GroupNorm (32 groups of 16 ch x 1024 px) -> f16, transposed ---
__global__ __launch_bounds__(256) void gn_kernel(const float* __restrict__ x,
                                                 const float* __restrict__ gamma,
                                                 const float* __restrict__ beta,
                                                 _Float16* __restrict__ hT) {
  const int b = blockIdx.x >> 5;
  const int g = blockIdx.x & 31;
  const int tid = threadIdx.x;
  const float* xg = x + ((size_t)(b * 512 + g * 16)) * 1024;
  __shared__ float s1[256], s2[256];
  float a = 0.f, q = 0.f;
  for (int i = tid; i < 16384; i += 256) { float v = xg[i]; a += v; q += v * v; }
  s1[tid] = a; s2[tid] = q;
  __syncthreads();
  for (int s = 128; s > 0; s >>= 1) {
    if (tid < s) { s1[tid] += s1[tid + s]; s2[tid] += s2[tid + s]; }
    __syncthreads();
  }
  const float mean = s1[0] * (1.f / 16384.f);
  const float var  = s2[0] * (1.f / 16384.f) - mean * mean;
  const float inv  = rsqrtf(var + 1e-5f);
  for (int i = tid; i < 16384; i += 256) {
    const int c = i >> 10, n = i & 1023;
    const int ch = g * 16 + c;
    const float v = (xg[i] - mean) * inv * gamma[ch] + beta[ch];
    hT[((size_t)(b * 1024 + n)) * 512 + ch] = (_Float16)v;
  }
}

// ---- Kernel 2: weight f32 -> f16 ---------------------------------------------
__global__ __launch_bounds__(256) void cvt_kernel(const float* __restrict__ wq,
                                                  const float* __restrict__ wp,
                                                  _Float16* __restrict__ wqh,
                                                  _Float16* __restrict__ wph) {
  const int i = blockIdx.x * 256 + threadIdx.x;
  if (i < 1536 * 512) wqh[i] = (_Float16)wq[i];
  if (i < 512 * 512)  wph[i] = (_Float16)wp[i];
}

// ---- Kernel 3: QKV GEMM (per batch: 1024 x 512 x 1536), head split -----------
__global__ __launch_bounds__(256) void qkv_kernel(const _Float16* __restrict__ hT,
                                                  const _Float16* __restrict__ wq,
                                                  const float* __restrict__ bias,
                                                  _Float16* __restrict__ Qh,
                                                  _Float16* __restrict__ Kh,
                                                  _Float16* __restrict__ Vt) {
  __shared__ alignas(16) _Float16 As[64 * 40];    // 64 rows x 32 + pad
  __shared__ alignas(16) _Float16 Bs[128 * 40];   // 128 rows x 32 + pad
  const int mt = blockIdx.x * 64;
  const int nt = blockIdx.y * 128;
  const int bat = blockIdx.z;
  const int tid = threadIdx.x, wave = tid >> 5, lane = tid & 31;
  const int wm = wave & 3, wn = wave >> 2;
  const int col = lane & 15, hf = lane >> 4;
  const _Float16* Ab = hT + (size_t)bat * 1024 * 512;
  const v8f zero8 = {0.f, 0.f, 0.f, 0.f, 0.f, 0.f, 0.f, 0.f};
  v8f acc[4];
#pragma unroll
  for (int t = 0; t < 4; ++t) acc[t] = zero8;

  for (int kc = 0; kc < 16; ++kc) {
    {   // A tile 64x32 halves: 256 lanes x 16B, async global->LDS
      const int row = tid >> 2, seg = tid & 3;
      ASYNC_G2L_B128(lds_off(&As[row * 40 + seg * 8]),
                     (unsigned)((mt + row) * 1024 + kc * 64 + seg * 16), Ab);
    }
#pragma unroll
    for (int r = 0; r < 2; ++r) {   // B tile 128x32 halves: 512 x 16B
      const int i = tid + r * 256;
      const int row = i >> 2, seg = i & 3;
      ASYNC_G2L_B128(lds_off(&Bs[row * 40 + seg * 8]),
                     (unsigned)((nt + row) * 1024 + kc * 64 + seg * 16), wq);
    }
    if (kc + 1 < 16) {              // global_prefetch_b8 next slab
      __builtin_prefetch(Ab + (size_t)(mt + (tid >> 2)) * 512 + (kc + 1) * 32, 0, 1);
      __builtin_prefetch(wq + (size_t)(nt + (tid >> 1)) * 512 + (kc + 1) * 32, 0, 1);
    }
    WAIT_ASYNC();
    __syncthreads();
    const v16h af = frag_a(&As[wm * 16 * 40], 40);
#pragma unroll
    for (int t = 0; t < 4; ++t) {
      const v16h bf = frag_b(&Bs[(wn * 64 + t * 16) * 40], 40);
      acc[t] = WMMA(af, bf, acc[t]);
    }
    __syncthreads();
  }

  const float qscale = 0.04419417382415922f;     // 512^-0.5 folded into Q
#pragma unroll
  for (int t = 0; t < 4; ++t) {
    const int o = nt + wn * 64 + t * 16 + col;
    const float bia = bias[o];
    const int head = o / 192, r = o - head * 192;   // per-head [q:64|k:64|v:64]
    const size_t hb = ((size_t)(bat * 8 + head)) << 16;  // *1024*64
#pragma unroll
    for (int j = 0; j < 8; ++j) {
      const int n = mt + wm * 16 + j + (hf << 3);
      const float v = acc[t][j] + bia;
      if (r < 64)       Qh[hb + (size_t)n * 64 + r] = (_Float16)(v * qscale);
      else if (r < 128) Kh[hb + (size_t)n * 64 + (r - 64)] = (_Float16)v;
      else              Vt[hb + (size_t)(r - 128) * 1024 + n] = (_Float16)v;
    }
  }
}

// ---- Kernel 4: flash attention. block = (qblock of 128, b*nh); 8 waves,
// each wave owns 16 query rows; loop 16 key-blocks of 64 with online softmax. --
__global__ __launch_bounds__(256) void attn_kernel(const _Float16* __restrict__ Qh,
                                                   const _Float16* __restrict__ Kh,
                                                   const _Float16* __restrict__ Vt,
                                                   _Float16* __restrict__ Oh) {
  __shared__ alignas(16) _Float16 Ks[64 * 72];       // [key][d], padded
  __shared__ alignas(16) _Float16 Vs[64 * 72];       // [d][key], padded
  __shared__ alignas(16) _Float16 Ps[8 * 16 * 72];   // per-wave P tile 16x64
  const int qb = blockIdx.x, bh = blockIdx.y;
  const int tid = threadIdx.x, wave = tid >> 5, lane = tid & 31;
  const int col = lane & 15, hf = lane >> 4;
  const _Float16* Qb = Qh + ((size_t)bh << 16) + (size_t)(qb * 128 + wave * 16) * 64;
  const _Float16* Kb = Kh + ((size_t)bh << 16);
  const _Float16* Vb = Vt + ((size_t)bh << 16);
  const v8f zero8 = {0.f, 0.f, 0.f, 0.f, 0.f, 0.f, 0.f, 0.f};

  const v16h q0 = frag_a(Qb, 64);        // d = 0..31
  const v16h q1 = frag_a(Qb + 32, 64);   // d = 32..63
  v8f acc[4];
  float mrow[8], lrow[8];
#pragma unroll
  for (int t = 0; t < 4; ++t) acc[t] = zero8;
#pragma unroll
  for (int j = 0; j < 8; ++j) { mrow[j] = -1e30f; lrow[j] = 0.f; }

  for (int kb = 0; kb < 16; ++kb) {
    __syncthreads();   // all waves done reading Ks/Vs from previous iteration
#pragma unroll
    for (int r = 0; r < 2; ++r) {        // K block 64x64 + V block 64x64 (async)
      const int i = tid + r * 256;
      const int row = i >> 3, seg = i & 7;
      ASYNC_G2L_B128(lds_off(&Ks[row * 72 + seg * 8]),
                     (unsigned)((kb * 64 + row) * 128 + seg * 16), Kb);
      ASYNC_G2L_B128(lds_off(&Vs[row * 72 + seg * 8]),
                     (unsigned)(row * 2048 + kb * 128 + seg * 16), Vb);
    }
    WAIT_ASYNC();
    __syncthreads();

    v8f S[4];
#pragma unroll
    for (int t = 0; t < 4; ++t) {                    // S = Q K^T (scale in Q)
      S[t] = zero8;
      S[t] = WMMA(q0, frag_b(&Ks[t * 16 * 72], 72), S[t]);
      S[t] = WMMA(q1, frag_b(&Ks[t * 16 * 72 + 32], 72), S[t]);
    }
    // online softmax: row stats live replicated across the 16 lanes of a half
#pragma unroll
    for (int j = 0; j < 8; ++j) {
      float mx = fmaxf(fmaxf(S[0][j], S[1][j]), fmaxf(S[2][j], S[3][j]));
#pragma unroll
      for (int d = 1; d < 16; d <<= 1) mx = fmaxf(mx, __shfl_xor(mx, d, 32));
      const float mn = fmaxf(mrow[j], mx);
      const float al = __expf(mrow[j] - mn);
      mrow[j] = mn;
      float sm = 0.f;
#pragma unroll
      for (int t = 0; t < 4; ++t) {
        const float p = __expf(S[t][j] - mn);
        S[t][j] = p;
        sm += p;
      }
#pragma unroll
      for (int d = 1; d < 16; d <<= 1) sm += __shfl_xor(sm, d, 32);
      lrow[j] = lrow[j] * al + sm;
#pragma unroll
      for (int t = 0; t < 4; ++t) acc[t][j] *= al;
    }
    // accumulator(C) layout -> A layout via per-wave LDS tile
    _Float16* Pw = &Ps[wave * 16 * 72];
#pragma unroll
    for (int t = 0; t < 4; ++t)
#pragma unroll
      for (int j = 0; j < 8; ++j)
        Pw[(j + (hf << 3)) * 72 + t * 16 + col] = (_Float16)S[t][j];
    const v16h p0 = frag_a(Pw, 72);        // keys 0..31
    const v16h p1 = frag_a(Pw + 32, 72);   // keys 32..63
#pragma unroll
    for (int t = 0; t < 4; ++t) {          // O += P V
      acc[t] = WMMA(p0, frag_b(&Vs[t * 16 * 72], 72), acc[t]);
      acc[t] = WMMA(p1, frag_b(&Vs[t * 16 * 72 + 32], 72), acc[t]);
    }
  }

  const int bq = bh >> 3, head = bh & 7;
#pragma unroll
  for (int t = 0; t < 4; ++t)
#pragma unroll
    for (int j = 0; j < 8; ++j) {
      const int n = qb * 128 + wave * 16 + j + (hf << 3);
      const float v = acc[t][j] / lrow[j];
      Oh[((size_t)(bq * 1024 + n)) * 512 + head * 64 + t * 16 + col] = (_Float16)v;
    }
}

// ---- Kernel 5: proj GEMM (1024 x 512 x 512) + bias + residual, fp32 out ------
__global__ __launch_bounds__(256) void proj_kernel(const _Float16* __restrict__ Oh,
                                                   const _Float16* __restrict__ wp,
                                                   const float* __restrict__ bias,
                                                   const float* __restrict__ x,
                                                   float* __restrict__ out) {
  __shared__ alignas(16) _Float16 As[64 * 40];
  __shared__ alignas(16) _Float16 Bs[128 * 40];
  const int mt = blockIdx.x * 64;
  const int nt = blockIdx.y * 128;
  const int bat = blockIdx.z;
  const int tid = threadIdx.x, wave = tid >> 5, lane = tid & 31;
  const int wm = wave & 3, wn = wave >> 2;
  const int col = lane & 15, hf = lane >> 4;
  const _Float16* Ab = Oh + (size_t)bat * 1024 * 512;
  const v8f zero8 = {0.f, 0.f, 0.f, 0.f, 0.f, 0.f, 0.f, 0.f};
  v8f acc[4];
#pragma unroll
  for (int t = 0; t < 4; ++t) acc[t] = zero8;

  for (int kc = 0; kc < 16; ++kc) {
    {
      const int row = tid >> 2, seg = tid & 3;
      ASYNC_G2L_B128(lds_off(&As[row * 40 + seg * 8]),
                     (unsigned)((mt + row) * 1024 + kc * 64 + seg * 16), Ab);
    }
#pragma unroll
    for (int r = 0; r < 2; ++r) {
      const int i = tid + r * 256;
      const int row = i >> 2, seg = i & 3;
      ASYNC_G2L_B128(lds_off(&Bs[row * 40 + seg * 8]),
                     (unsigned)((nt + row) * 1024 + kc * 64 + seg * 16), wp);
    }
    if (kc + 1 < 16) {
      __builtin_prefetch(Ab + (size_t)(mt + (tid >> 2)) * 512 + (kc + 1) * 32, 0, 1);
      __builtin_prefetch(wp + (size_t)(nt + (tid >> 1)) * 512 + (kc + 1) * 32, 0, 1);
    }
    WAIT_ASYNC();
    __syncthreads();
    const v16h af = frag_a(&As[wm * 16 * 40], 40);
#pragma unroll
    for (int t = 0; t < 4; ++t) {
      const v16h bf = frag_b(&Bs[(wn * 64 + t * 16) * 40], 40);
      acc[t] = WMMA(af, bf, acc[t]);
    }
    __syncthreads();
  }

#pragma unroll
  for (int t = 0; t < 4; ++t) {
    const int o = nt + wn * 64 + t * 16 + col;
    const float bia = bias[o];
#pragma unroll
    for (int j = 0; j < 8; ++j) {
      const int n = mt + wm * 16 + j + (hf << 3);
      const size_t idx = ((size_t)(bat * 512 + o)) * 1024 + n;
      out[idx] = acc[t][j] + bia + x[idx];   // residual
    }
  }
}

extern "C" void kernel_launch(void* const* d_in, const int* in_sizes, int n_in,
                              void* d_out, int out_size, void* d_ws, size_t ws_size,
                              hipStream_t stream) {
  (void)in_sizes; (void)n_in; (void)out_size; (void)ws_size;
  const float* x      = (const float*)d_in[0];
  const float* gamma  = (const float*)d_in[1];
  const float* beta   = (const float*)d_in[2];
  const float* w_qkv  = (const float*)d_in[3];
  const float* b_qkv  = (const float*)d_in[4];
  const float* w_proj = (const float*)d_in[5];
  const float* b_proj = (const float*)d_in[6];
  float* out = (float*)d_out;

  const size_t SZ = (size_t)16 * 1024 * 1024;   // one f16 [B][HW][C] tensor
  char* w = (char*)d_ws;
  _Float16* hT  = (_Float16*)(w);
  _Float16* Qh  = (_Float16*)(w + SZ);
  _Float16* Kh  = (_Float16*)(w + 2 * SZ);
  _Float16* Vt  = (_Float16*)(w + 3 * SZ);
  _Float16* Oh  = (_Float16*)(w + 4 * SZ);
  _Float16* wqh = (_Float16*)(w + 5 * SZ);
  _Float16* wph = (_Float16*)(w + 5 * SZ + (size_t)1536 * 512 * 2);

  cvt_kernel<<<3072, 256, 0, stream>>>(w_qkv, w_proj, wqh, wph);
  gn_kernel<<<512, 256, 0, stream>>>(x, gamma, beta, hT);
  qkv_kernel<<<dim3(16, 12, 16), 256, 0, stream>>>(hT, wqh, b_qkv, Qh, Kh, Vt);
  attn_kernel<<<dim3(8, 128), 256, 0, stream>>>(Qh, Kh, Vt, Oh);
  proj_kernel<<<dim3(16, 4, 16), 256, 0, stream>>>(Oh, wph, b_proj, x, out);
}
